// WordRNN_2723009266425
// MI455X (gfx1250) — matrix operand order repaired and checked by reading, not compiled, not run on hardware
//
#include <hip/hip_runtime.h>
#include <hip/hip_bf16.h>

// ---------------------------------------------------------------------------
// GRU word-RNN for MI455X (gfx1250): bf16 WMMA GEMMs, fp32 accumulation.
// Per-step fused kernel: [x_t ; h] @ [Wk ; Wr] with 4 accumulator sets
// (z, r, h_x, h_h) + gate nonlinearity + state update in the epilogue.
// Round 3: software-pipelined double-buffered LDS staging using
// GLOBAL_LOAD_ASYNC_TO_LDS_B128 (in-order ASYNCcnt completion lets us wait
// with s_wait_asynccnt 7 while the next tile's group is still in flight).
// ---------------------------------------------------------------------------

typedef __bf16 bf16_t;
typedef __attribute__((ext_vector_type(16))) __bf16 v16bf;
typedef __attribute__((ext_vector_type(8)))  __bf16 v8bf;
typedef __attribute__((ext_vector_type(8)))  float  v8f;

#define BATCH 4096
#define TSEQ  100
#define EMB   50
#define EPAD  64
#define HID   1024
#define H3    3072
#define NLAB  15
#define NTILE 34          // 2 x-phase K-tiles (EPAD/32) + 32 h-phase K-tiles

static __device__ __forceinline__ bf16_t f2bf(float f) {
  union { float f; unsigned u; } in; in.f = f;
  unsigned u = in.u;
  unsigned lsb = (u >> 16) & 1u;
  u += 0x7fffu + lsb;                       // round to nearest even
  unsigned short h = (unsigned short)(u >> 16);
  union { unsigned short s; bf16_t b; } out; out.s = h;
  return out.b;
}

static __device__ __forceinline__ float sigmoidf_(float x) {
  return 1.0f / (1.0f + __expf(-x));
}

// Async copy 16B/lane global->LDS (GLOBAL_LOAD_ASYNC_TO_LDS_B128, ASYNCcnt).
static __device__ __forceinline__ void async_cp_b128(const void* gsrc, void* ldst) {
  unsigned loff = (unsigned)(size_t)ldst;   // LDS byte offset = addr[31:0]
  asm volatile("global_load_async_to_lds_b128 %0, %1, off"
               :: "v"(loff), "v"(gsrc)
               : "memory");
}

static __device__ __forceinline__ void wait_async_le7() {
  asm volatile("s_wait_asynccnt 0x7" ::: "memory");
}
static __device__ __forceinline__ void wait_async_le0() {
  asm volatile("s_wait_asynccnt 0x0" ::: "memory");
}

// ---------------------------------------------------------------------------
// Pre-pass kernels
// ---------------------------------------------------------------------------
__global__ void gather_emb_kernel(const int* __restrict__ x,
                                  const float* __restrict__ tab,
                                  bf16_t* __restrict__ embT) {
  const unsigned idx = blockIdx.x * 256u + threadIdx.x;   // < T*B*EPAD
  const unsigned t   = idx / (BATCH * EPAD);
  const unsigned rem = idx - t * (BATCH * EPAD);
  const unsigned b   = rem >> 6;
  const unsigned e   = rem & 63u;
  const int tok = x[b * TSEQ + t];
  float v = (e < EMB) ? tab[(size_t)tok * EMB + e] : 0.0f;
  embT[idx] = f2bf(v);
}

__global__ void conv_wk_kernel(const float* __restrict__ w, bf16_t* __restrict__ wt) {
  const unsigned idx = blockIdx.x * 256u + threadIdx.x;   // < H3*EPAD
  const unsigned col = idx >> 6;
  const unsigned k   = idx & 63u;
  float v = (k < EMB) ? w[(size_t)k * H3 + col] : 0.0f;
  wt[idx] = f2bf(v);
}

__global__ void conv_wr_kernel(const float* __restrict__ w, bf16_t* __restrict__ wt) {
  const unsigned idx = blockIdx.x * 256u + threadIdx.x;   // < H3*HID
  const unsigned col = idx >> 10;
  const unsigned k   = idx & 1023u;
  wt[idx] = f2bf(w[(size_t)k * H3 + col]);
}

__global__ void init_h_kernel(float* __restrict__ hF, bf16_t* __restrict__ hB) {
  const unsigned idx = blockIdx.x * 256u + threadIdx.x;   // < BATCH*HID
  hF[idx] = 0.0f;
  hB[idx] = f2bf(0.0f);
}

// ---------------------------------------------------------------------------
// Compute one staged 32-K tile: 12 WMMAs per wave (2 mt x 2 nt x 3 gates).
// PH0 selects the h-gate accumulator set (x-part vs h-part).
// ---------------------------------------------------------------------------
template <bool PH0>
static __device__ __forceinline__ void compute_tile(
    const bf16_t* __restrict__ sAd,      // [64][32]
    const bf16_t* __restrict__ sBd,      // [3][128][32]
    int waveM, int waveN, int lane15, int kHalfA, int kHalfB,
    v8f (&acc)[4][2][2])
{
  v16bf af[2];
#pragma unroll
  for (int mt = 0; mt < 2; ++mt) {
    const int ar = waveM * 32 + mt * 16 + lane15;
    v8bf alo = *(const v8bf*)&sAd[ar * 32 + kHalfA];
    v8bf ahi = *(const v8bf*)&sAd[ar * 32 + 16 + kHalfA];
    af[mt] = __builtin_shufflevector(alo, ahi,
               0,1,2,3,4,5,6,7,8,9,10,11,12,13,14,15);
  }
#pragma unroll
  for (int nt = 0; nt < 2; ++nt) {
    const int nRow = waveN * 32 + nt * 16 + lane15;
#pragma unroll
    for (int g = 0; g < 3; ++g) {
      v8bf blo = *(const v8bf*)&sBd[g * (128 * 32) + nRow * 32 + kHalfB];
      v8bf bhi = *(const v8bf*)&sBd[g * (128 * 32) + nRow * 32 + kHalfB + 16];
      v16bf bfr = __builtin_shufflevector(blo, bhi,
                    0,1,2,3,4,5,6,7,8,9,10,11,12,13,14,15);
      const int aid = (g < 2) ? g : (PH0 ? 2 : 3);
#pragma unroll
      for (int mt = 0; mt < 2; ++mt) {
        acc[aid][mt][nt] = __builtin_amdgcn_wmma_f32_16x16x32_bf16(
            false, af[mt], false, bfr, (short)0, acc[aid][mt][nt],
            false, false);
      }
    }
  }
}

// ---------------------------------------------------------------------------
// One GRU timestep. Grid: (HID/128, BATCH/64), 256 threads = 8 waves (2M x 4N).
// ---------------------------------------------------------------------------
__global__ __launch_bounds__(256)
void gru_step_kernel(const bf16_t* __restrict__ embA,   // [BATCH][EPAD]
                     const bf16_t* __restrict__ hBF,    // [BATCH][HID]
                     const float*  __restrict__ hFin,   // [BATCH][HID]
                     const bf16_t* __restrict__ WkT,    // [H3][EPAD]
                     const bf16_t* __restrict__ WrT,    // [H3][HID]
                     const float*  __restrict__ bi,     // [H3]
                     const float*  __restrict__ brb,    // [H3]
                     float*        __restrict__ hFout,  // [BATCH][HID]
                     bf16_t*       __restrict__ hBout)  // [BATCH][HID]
{
  __shared__ alignas(16) bf16_t sA[2][64 * 32];         // double-buffered
  __shared__ alignas(16) bf16_t sB[2][3 * 128 * 32];

  const int tid   = threadIdx.x;
  const int lane  = tid & 31;
  const int wid   = tid >> 5;
  const int waveM = wid >> 2;            // 0..1
  const int waveN = wid & 3;             // 0..3
  const int rowBase = blockIdx.y * 64;
  const int colBase = blockIdx.x * 128;

  v8f acc[4][2][2];
#pragma unroll
  for (int s = 0; s < 4; ++s)
#pragma unroll
    for (int mt = 0; mt < 2; ++mt)
#pragma unroll
      for (int nt = 0; nt < 2; ++nt)
#pragma unroll
        for (int i = 0; i < 8; ++i) acc[s][mt][nt][i] = 0.0f;

  // staging indices (7 async b128 per thread per tile)
  const int aRowLd = tid >> 2;           // 0..63
  const int aColLd = (tid & 3) * 8;

  // fragment lane indices (ISA 16-bit WMMA layouts)
  const int kHalfA = (lane >> 4) * 8;
  const int kHalfB = (lane >> 4) * 8;
  const int lane15 = lane & 15;

  // Issue the 7 async loads for K-tile t into LDS buffer d.
  auto stage = [&](int t, int d) {
    const bf16_t* Ag;
    const bf16_t* Wt;
    int strd, k0;
    if (t < 2) { Ag = embA; Wt = WkT; strd = EPAD; k0 = t * 32; }
    else       { Ag = hBF;  Wt = WrT; strd = HID;  k0 = (t - 2) * 32; }
    async_cp_b128(Ag + (size_t)(rowBase + aRowLd) * strd + k0 + aColLd,
                  &sA[d][aRowLd * 32 + aColLd]);
#pragma unroll
    for (int g = 0; g < 3; ++g) {
#pragma unroll
      for (int j = 0; j < 2; ++j) {
        const int unit = tid + 256 * j;          // 0..511
        const int nrow = unit >> 2;              // 0..127
        const int cgrp = (unit & 3) * 8;
        async_cp_b128(Wt + (size_t)(g * HID + colBase + nrow) * strd + k0 + cgrp,
                      &sB[d][g * (128 * 32) + nrow * 32 + cgrp]);
      }
    }
  };

  // ---- software-pipelined K loop over 34 tiles ----
  stage(0, 0);
  for (int t = 0; t < NTILE; ++t) {
    const int d = t & 1;
    if (t + 1 < NTILE) {
      stage(t + 1, d ^ 1);     // prefetch next tile into the other buffer
      wait_async_le7();        // tile t's group (7 asyncs) has landed
    } else {
      wait_async_le0();
    }
    __syncthreads();           // all waves' staging for tile t visible
    if (t < 2) {
      compute_tile<true >(sA[d], sB[d], waveM, waveN, lane15, kHalfA, kHalfB, acc);
    } else {
      compute_tile<false>(sA[d], sB[d], waveM, waveN, lane15, kHalfA, kHalfB, acc);
    }
    __syncthreads();           // buffer d safe to re-stage at iteration t+1
  }

  // ---- epilogue: gates + state update ----
#pragma unroll
  for (int nt = 0; nt < 2; ++nt) {
    const int col = colBase + waveN * 32 + nt * 16 + lane15;
    const float bz  = bi[col]           + brb[col];
    const float brg = bi[HID + col]     + brb[HID + col];
    const float bx  = bi[2 * HID + col];
    const float bh  = brb[2 * HID + col];
#pragma unroll
    for (int mt = 0; mt < 2; ++mt) {
#pragma unroll
      for (int v = 0; v < 8; ++v) {
        const int row = rowBase + waveM * 32 + mt * 16 + (lane >> 4) * 8 + v;
        const float z  = sigmoidf_(acc[0][mt][nt][v] + bz);
        const float r  = sigmoidf_(acc[1][mt][nt][v] + brg);
        const float hc = tanhf(acc[2][mt][nt][v] + bx +
                               r * (acc[3][mt][nt][v] + bh));
        const float hp = hFin[(size_t)row * HID + col];
        const float hn = z * hp + (1.0f - z) * hc;
        hFout[(size_t)row * HID + col] = hn;
        hBout[(size_t)row * HID + col] = f2bf(hn);
      }
    }
  }
}

// ---------------------------------------------------------------------------
// Final projection: logits[B,15] = h @ Wd + bd  (tiny; scalar kernel)
// ---------------------------------------------------------------------------
__global__ void logits_kernel(const float* __restrict__ h,
                              const float* __restrict__ Wd,
                              const float* __restrict__ bd,
                              float* __restrict__ out) {
  const int idx = blockIdx.x * 256 + threadIdx.x;
  if (idx >= BATCH * NLAB) return;
  const int row = idx / NLAB;
  const int col = idx - row * NLAB;
  float a = 0.0f;
  const float* hr = h + (size_t)row * HID;
  for (int k = 0; k < HID; ++k) a += hr[k] * Wd[(size_t)k * NLAB + col];
  out[idx] = a + bd[col];
}

// ---------------------------------------------------------------------------
extern "C" void kernel_launch(void* const* d_in, const int* in_sizes, int n_in,
                              void* d_out, int out_size, void* d_ws, size_t ws_size,
                              hipStream_t stream) {
  (void)in_sizes; (void)n_in; (void)out_size; (void)ws_size;

  const int*   x         = (const int*)  d_in[0];
  /* d_in[1] = drop_rate (0, static) — unused */
  const float* emb_table = (const float*)d_in[2];
  const float* kern      = (const float*)d_in[3];
  const float* reck      = (const float*)d_in[4];
  const float* bi        = (const float*)d_in[5];
  const float* br        = (const float*)d_in[6];
  const float* Wd        = (const float*)d_in[7];
  const float* bd        = (const float*)d_in[8];
  float*       out       = (float*)d_out;

  // workspace layout
  char* ws = (char*)d_ws;
  size_t off = 0;
  bf16_t* embT = (bf16_t*)(ws + off); off += (size_t)TSEQ * BATCH * EPAD * sizeof(bf16_t);
  bf16_t* WkT  = (bf16_t*)(ws + off); off += (size_t)H3 * EPAD * sizeof(bf16_t);
  bf16_t* WrT  = (bf16_t*)(ws + off); off += (size_t)H3 * HID * sizeof(bf16_t);
  float*  hF[2];
  hF[0] = (float*)(ws + off); off += (size_t)BATCH * HID * sizeof(float);
  hF[1] = (float*)(ws + off); off += (size_t)BATCH * HID * sizeof(float);
  bf16_t* hB[2];
  hB[0] = (bf16_t*)(ws + off); off += (size_t)BATCH * HID * sizeof(bf16_t);
  hB[1] = (bf16_t*)(ws + off); off += (size_t)BATCH * HID * sizeof(bf16_t);

  // pre-pass
  gather_emb_kernel<<<(TSEQ * BATCH * EPAD) / 256, 256, 0, stream>>>(x, emb_table, embT);
  conv_wk_kernel<<<(H3 * EPAD) / 256, 256, 0, stream>>>(kern, WkT);
  conv_wr_kernel<<<(H3 * HID) / 256, 256, 0, stream>>>(reck, WrT);
  init_h_kernel<<<(BATCH * HID) / 256, 256, 0, stream>>>(hF[0], hB[0]);

  // sequential GRU scan
  int cur = 0;
  dim3 grid(HID / 128, BATCH / 64);
  for (int t = 0; t < TSEQ; ++t) {
    gru_step_kernel<<<grid, 256, 0, stream>>>(
        embT + (size_t)t * BATCH * EPAD,
        hB[cur], hF[cur], WkT, WrT, bi, br,
        hF[cur ^ 1], hB[cur ^ 1]);
    cur ^= 1;
  }

  logits_kernel<<<(BATCH * NLAB + 255) / 256, 256, 0, stream>>>(hF[cur], Wd, bd, out);
}
